// MultiHeadLatentAttention_17927193493795
// MI455X (gfx1250) — matrix operand order repaired and checked by reading, not compiled, not run on hardware
//
#include <hip/hip_runtime.h>

// ---------------------------------------------------------------------------
// MLA attention for MI455X (gfx1250, wave32, WMMA).
// Pipeline: q = hs@Wq ; kv_c = sig(1)*hs@Wc ; kv = kv_c@Wup ;
//           o = softmax(q kv^T * D^-1/2) kv ; out = o@Wo
// Matmuls on v_wmma_f32_16x16x32_f16. K/V tiles double-buffered in LDS via
// global_load_async_to_lds_b128 (ASYNCcnt). GEMMs software-pipeline global
// loads through registers. f16 intermediates in d_ws.
// ---------------------------------------------------------------------------

typedef __attribute__((ext_vector_type(16))) _Float16 v16h;
typedef __attribute__((ext_vector_type(8)))  float    v8f;

#define WMMA_F16(a, b, c) \
  __builtin_amdgcn_wmma_f32_16x16x32_f16(false, (a), false, (b), (short)0, (c), false, false)

constexpr int   kB    = 4;
constexpr int   kS    = 2048;
constexpr int   kHID  = 2048;
constexpr int   kH    = 16;
constexpr int   kHKV  = 4;
constexpr int   kD    = 128;
constexpr int   kRANK = 512;
constexpr float kScale = 0.088388347648318447f;  // D^-0.5
constexpr float kSig1  = 0.7310585786300049f;    // sigmoid(1.0)

// ---------------------------------------------------------------------------
// Tiled GEMM: C[M,N] = scale * A[M,K] @ B[K,N]
// 256 threads / 8 waves; block tile 128(M) x 128(N); each wave owns 32x64
// -> 8 WMMA per 32-wide k-step. Global loads for tile k+1 are issued into
// registers while WMMAs consume tile k from LDS (software pipeline).
// OUTMODE 0: row-major MxN. OUTMODE 1: head-split (B, N/kD, kS, kD).
// ---------------------------------------------------------------------------
template <typename AT, typename CT, int OUTMODE>
__global__ __launch_bounds__(256) void gemm_wmma(
    const AT* __restrict__ A, const float* __restrict__ Bm, CT* __restrict__ C,
    int M, int N, int K, float scale) {
  __shared__ __align__(16) _Float16 lA[128][40];   // 128 rows x 32 k (+pad)
  __shared__ __align__(16) _Float16 lB[32][136];   // 32 k x 128 n (+pad)

  const int t    = threadIdx.x;
  const int lane = t & 31;
  const int wave = t >> 5;
  const int wm   = wave & 3;   // 4 waves along M (32 rows each)
  const int wn   = wave >> 2;  // 2 waves along N (64 cols each)
  const int m0   = blockIdx.y * 128;
  const int n0   = blockIdx.x * 128;

  // Per-thread contiguous staging assignments (16 elems each for A and B).
  const int ar = t >> 1;         // A: 2 threads per 32-wide row
  const int ac = (t & 1) * 16;
  const int br = t >> 3;         // B: 8 threads per 128-wide row
  const int bc = (t & 7) * 16;

  const AT*    aSrc = A + (size_t)(m0 + ar) * K + ac;
  const float* bSrc = Bm + (size_t)br * N + (n0 + bc);

  v8f acc[2][4] = {};

  // Register staging for the pipelined tile.
  float4 axf[4];  // AT == float path
  v16h   axh;     // AT == _Float16 path
  float4 bxf[4];

  // Prologue: load tile k0 = 0.
  if constexpr (sizeof(AT) == 4) {
    const float4* v4 = (const float4*)aSrc;
#pragma unroll
    for (int j = 0; j < 4; ++j) axf[j] = v4[j];
  } else {
    axh = *(const v16h*)aSrc;
  }
  {
    const float4* v4 = (const float4*)bSrc;
#pragma unroll
    for (int j = 0; j < 4; ++j) bxf[j] = v4[j];
  }

  for (int k0 = 0; k0 < K; k0 += 32) {
    // Commit staged registers to LDS (f32 -> f16 conversion here).
    if constexpr (sizeof(AT) == 4) {
#pragma unroll
      for (int j = 0; j < 4; ++j) {
        lA[ar][ac + 4 * j + 0] = (_Float16)axf[j].x;
        lA[ar][ac + 4 * j + 1] = (_Float16)axf[j].y;
        lA[ar][ac + 4 * j + 2] = (_Float16)axf[j].z;
        lA[ar][ac + 4 * j + 3] = (_Float16)axf[j].w;
      }
    } else {
#pragma unroll
      for (int j = 0; j < 16; ++j) lA[ar][ac + j] = axh[j];
    }
#pragma unroll
    for (int j = 0; j < 4; ++j) {
      lB[br][bc + 4 * j + 0] = (_Float16)bxf[j].x;
      lB[br][bc + 4 * j + 1] = (_Float16)bxf[j].y;
      lB[br][bc + 4 * j + 2] = (_Float16)bxf[j].z;
      lB[br][bc + 4 * j + 3] = (_Float16)bxf[j].w;
    }
    __syncthreads();

    // Issue next tile's global loads; results consumed next iteration, so
    // their latency hides behind this iteration's WMMAs.
    if (k0 + 32 < K) {
      if constexpr (sizeof(AT) == 4) {
        const float4* v4 = (const float4*)(aSrc + (k0 + 32));
#pragma unroll
        for (int j = 0; j < 4; ++j) axf[j] = v4[j];
      } else {
        axh = *(const v16h*)(aSrc + (k0 + 32));
      }
      const float4* v4 = (const float4*)(bSrc + (size_t)(k0 + 32) * N);
#pragma unroll
      for (int j = 0; j < 4; ++j) bxf[j] = v4[j];
    }

    // WMMA fragments from LDS (contiguous k = 16*(lane/16) + i in both A and B).
    v16h af[2], bf[4];
    const int amr = wm * 32 + (lane & 15);
    const int ak  = (lane >> 4) * 16;
#pragma unroll
    for (int i = 0; i < 16; ++i) {
      int kk   = ak + i;
      af[0][i] = lA[amr][kk];
      af[1][i] = lA[amr + 16][kk];
    }
    const int bn = wn * 64 + (lane & 15);
    const int bk = (lane >> 4) * 16;
#pragma unroll
    for (int i = 0; i < 16; ++i) {
      bf[0][i] = lB[bk + i][bn];
      bf[1][i] = lB[bk + i][bn + 16];
      bf[2][i] = lB[bk + i][bn + 32];
      bf[3][i] = lB[bk + i][bn + 48];
    }

#pragma unroll
    for (int mi = 0; mi < 2; ++mi)
#pragma unroll
      for (int ni = 0; ni < 4; ++ni)
        acc[mi][ni] = WMMA_F16(af[mi], bf[ni], acc[mi][ni]);

    __syncthreads();
  }

  // Store the eight 16x16 tiles.
  const int colIn = lane & 15;
  const int rowIn = (lane >> 4) * 8;
#pragma unroll
  for (int mi = 0; mi < 2; ++mi) {
#pragma unroll
    for (int ni = 0; ni < 4; ++ni) {
      const int mbase = m0 + wm * 32 + mi * 16 + rowIn;
      const int nbase = n0 + wn * 64 + ni * 16 + colIn;
#pragma unroll
      for (int r = 0; r < 8; ++r) {
        int   mm = mbase + r;
        float v  = acc[mi][ni][r] * scale;
        size_t off;
        if constexpr (OUTMODE == 0) {
          off = (size_t)mm * N + nbase;
        } else {  // (B, N/kD heads, kS, kD) -- constant divisors -> shifts
          int b  = mm / kS, s = mm - b * kS;
          int h  = nbase / kD, d = nbase - h * kD;
          int Hn = N / kD;
          off = (((size_t)b * Hn + h) * kS + s) * (size_t)kD + d;
        }
        C[off] = (CT)v;
      }
    }
  }
}

// ---------------------------------------------------------------------------
// Flash attention (K == V == decompressed kv). One workgroup per
// (b, h, 128-row q block); 8 waves each own 16 q rows. 32-key x 128-dim K/V
// blocks double-buffered in LDS via global_load_async_to_lds_b128: block i+1
// is in flight while block i is consumed by QK^T and PV WMMAs.
// ---------------------------------------------------------------------------
__global__ __launch_bounds__(256) void mla_attention(
    const _Float16* __restrict__ Q, const _Float16* __restrict__ KV,
    _Float16* __restrict__ Out) {
  __shared__ __align__(16) _Float16 lKV[2][32][kD + 8];  // double buffer
  __shared__ __align__(16) _Float16 lP[8][16][40];       // per-wave P bounce

  const int t    = threadIdx.x;
  const int lane = t & 31;
  const int wave = t >> 5;
  const int bid  = blockIdx.x;
  const int qblk = bid & 15;         // S/128 = 16 q blocks
  const int h    = (bid >> 4) & 15;  // 16 heads
  const int b    = bid >> 8;         // 4 batches
  const int hkv  = h >> 2;           // H/HKV = 4
  const int qbase = qblk * 128 + wave * 16;

  const _Float16* qptr  = Q + (((size_t)(b * kH + h) * kS) + qbase) * kD;
  const _Float16* kvptr = KV + ((size_t)(b * kHKV + hkv) * kS) * kD;

  // Async staging: 32 rows x 256B per buffer; 16B per lane-transfer,
  // 512 transfers / 256 threads = 2 per thread.
  constexpr unsigned kBufBytes = 32u * (kD + 8) * 2;  // 8704
  unsigned ldsDst0[2], gOffBase[2];
#pragma unroll
  for (int j = 0; j < 2; ++j) {
    int e  = t + j * 256;
    int r  = e >> 4;        // key row 0..31
    int c8 = (e & 15) * 8;  // half index within row (16B chunks)
    ldsDst0[j]  = (unsigned)(size_t)&lKV[0][r][c8];
    gOffBase[j] = (unsigned)((r * kD + c8) * 2);  // bytes
  }

  // Q fragments (4 chunks of K=32 along D), pre-scaled by D^-0.5.
  v16h qf[4];
  {
    const int qm = lane & 15;
    const int qk = (lane >> 4) * 16;
#pragma unroll
    for (int c = 0; c < 4; ++c) {
#pragma unroll
      for (int i = 0; i < 16; ++i) {
        int d    = c * 32 + qk + i;
        qf[c][i] = (_Float16)((float)qptr[(size_t)qm * kD + d] * kScale);
      }
    }
  }

  v8f   oacc[8] = {};  // 16 q rows x 128 dims (8 d-tiles)
  float mrow[8], lrow[8];
#pragma unroll
  for (int r = 0; r < 8; ++r) { mrow[r] = -1e30f; lrow[r] = 0.0f; }

  // Prologue: async-fill buffer 0 with the first key block.
#pragma unroll
  for (int j = 0; j < 2; ++j) {
    asm volatile("global_load_async_to_lds_b128 %0, %1, %2"
                 :: "v"(ldsDst0[j]), "v"(gOffBase[j]), "s"(kvptr)
                 : "memory");
  }
  asm volatile("s_wait_asynccnt 0x0" ::: "memory");
  __syncthreads();

  for (int kb = 0; kb < kS; kb += 32) {
    const int buf = (kb >> 5) & 1;

    // Kick off the async copy of the NEXT key block into the other buffer;
    // its latency hides behind this block's 16 WMMAs + softmax.
    if (kb + 32 < kS) {
      const unsigned nextOff = (unsigned)((kb + 32) * kD * 2);
      const unsigned bufOff  = (buf ^ 1) ? kBufBytes : 0u;
#pragma unroll
      for (int j = 0; j < 2; ++j) {
        unsigned lds  = ldsDst0[j] + bufOff;
        unsigned voff = gOffBase[j] + nextOff;
        asm volatile("global_load_async_to_lds_b128 %0, %1, %2"
                     :: "v"(lds), "v"(voff), "s"(kvptr)
                     : "memory");
      }
    }

    // Scores: 16 q x 32 keys via 8 WMMAs (4 D-chunks x 2 key halves).
    v8f s0 = {}, s1 = {};
    {
      const int key = lane & 15;
      const int kk  = (lane >> 4) * 16;
#pragma unroll
      for (int c = 0; c < 4; ++c) {
        v16h kb0, kb1;
#pragma unroll
        for (int i = 0; i < 16; ++i) {
          int d  = c * 32 + kk + i;
          kb0[i] = lKV[buf][key][d];
          kb1[i] = lKV[buf][key + 16][d];
        }
        s0 = WMMA_F16(qf[c], kb0, s0);
        s1 = WMMA_F16(qf[c], kb1, s1);
      }
    }

    // Online softmax. Row r of this lane = r + (lane>=16 ? 8 : 0); 16-lane
    // halves share rows -> xor-butterfly (masks 1/2/4/8 stay in-half).
#pragma unroll
    for (int r = 0; r < 8; ++r) {
      float v = fmaxf(s0[r], s1[r]);
#pragma unroll
      for (int x = 1; x < 16; x <<= 1) v = fmaxf(v, __shfl_xor(v, x, 32));
      float mnew = fmaxf(mrow[r], v);
      float corr = __expf(mrow[r] - mnew);
      float p0   = __expf(s0[r] - mnew);
      float p1   = __expf(s1[r] - mnew);
      s0[r] = p0;
      s1[r] = p1;
      float ps = p0 + p1;
#pragma unroll
      for (int x = 1; x < 16; x <<= 1) ps += __shfl_xor(ps, x, 32);
      lrow[r] = lrow[r] * corr + ps;
      mrow[r] = mnew;
#pragma unroll
      for (int dt = 0; dt < 8; ++dt) oacc[dt][r] *= corr;
    }

    // Transpose P from C-layout to A-layout through per-wave LDS bounce.
    {
      const int pn = lane & 15;
      const int mb = (lane >> 4) * 8;
#pragma unroll
      for (int r = 0; r < 8; ++r) {
        lP[wave][mb + r][pn]      = (_Float16)s0[r];
        lP[wave][mb + r][pn + 16] = (_Float16)s1[r];
      }
    }
    v16h pf;
    {
      const int pm = lane & 15;
      const int pk = (lane >> 4) * 16;
#pragma unroll
      for (int i = 0; i < 16; ++i) {
        int k = pk + i;
        pf[i] = lP[wave][pm][k];
      }
    }

    // PV: 8 WMMAs accumulate 16x128 output.
    {
      const int vk = (lane >> 4) * 16;
#pragma unroll
      for (int dt = 0; dt < 8; ++dt) {
        v16h vf;
        const int vn = dt * 16 + (lane & 15);
#pragma unroll
        for (int i = 0; i < 16; ++i) vf[i] = lKV[buf][vk + i][vn];
        oacc[dt] = WMMA_F16(pf, vf, oacc[dt]);
      }
    }

    // Drain this wave's async copy of the next block, then workgroup-sync so
    // everyone may read it (and nobody still reads the buffer we refill next).
    asm volatile("s_wait_asynccnt 0x0" ::: "memory");
    __syncthreads();
  }

  // Normalize and store as (B, S, H*D) f16 feeding the Wo GEMM.
  const int orow = (lane >> 4) * 8;
  const int ocol = lane & 15;
#pragma unroll
  for (int dt = 0; dt < 8; ++dt) {
#pragma unroll
    for (int r = 0; r < 8; ++r) {
      int    s   = qbase + orow + r;
      int    d   = dt * 16 + ocol;
      float  v   = oacc[dt][r] / lrow[r];
      size_t off = ((size_t)(b * kS + s)) * (kH * kD) + h * kD + d;
      Out[off] = (_Float16)v;
    }
  }
}

// ---------------------------------------------------------------------------
extern "C" void kernel_launch(void* const* d_in, const int* in_sizes, int n_in,
                              void* d_out, int out_size, void* d_ws, size_t ws_size,
                              hipStream_t stream) {
  const float* hs  = (const float*)d_in[0];
  const float* Wq  = (const float*)d_in[1];
  const float* Wc  = (const float*)d_in[2];
  const float* Wup = (const float*)d_in[3];
  const float* Wo  = (const float*)d_in[4];
  float*       out = (float*)d_out;

  // Workspace layout (f16): q | kv_c | kv | attn_out  (~84 MB total)
  _Float16* q   = (_Float16*)d_ws;
  _Float16* kvc = q + (size_t)kB * kH * kS * kD;
  _Float16* kv  = kvc + (size_t)kB * kS * kRANK;
  _Float16* ao  = kv + (size_t)kB * kS * kHKV * kD;

  const int M = kB * kS;  // 8192
  dim3 blk(256);

  // 1) Q projection -> (B,H,S,D) f16
  gemm_wmma<float, _Float16, 1><<<dim3((kH * kD) / 128, M / 128), blk, 0, stream>>>(
      hs, Wq, q, M, kH * kD, kHID, 1.0f);

  // 2) Low-rank compression + sigmoid(1) gate
  gemm_wmma<float, _Float16, 0><<<dim3(kRANK / 128, M / 128), blk, 0, stream>>>(
      hs, Wc, kvc, M, kRANK, kHID, kSig1);

  // 3) Decompression -> (B,HKV,S,D) f16
  gemm_wmma<_Float16, _Float16, 1><<<dim3((kHKV * kD) / 128, M / 128), blk, 0, stream>>>(
      kvc, Wup, kv, M, kHKV * kD, kRANK, 1.0f);

  // 4) Flash attention: 4*16*16 = 1024 workgroups
  mla_attention<<<dim3(kB * kH * (kS / 128)), blk, 0, stream>>>(q, kv, ao);

  // 5) Output projection -> f32 d_out
  gemm_wmma<_Float16, float, 0><<<dim3(kHID / 128, M / 128), blk, 0, stream>>>(
      ao, Wo, out, M, kHID, kH * kD, 1.0f);
}
